// VMFLoss_18622978195459
// MI455X (gfx1250) — matrix operand order repaired and checked by reading, not compiled, not run on hardware
//
#include <hip/hip_runtime.h>
#include <math.h>

typedef __attribute__((ext_vector_type(16))) _Float16 v16h;
typedef __attribute__((ext_vector_type(8)))  _Float16 v8h;
typedef __attribute__((ext_vector_type(2)))  _Float16 v2h;
typedef __attribute__((ext_vector_type(8)))  float    v8f;

#define BATCH 8192
#define DIM   64
#define TILE  128
#define NB    (BATCH / TILE)            // 64 tile-blocks per side
#define NPAIR (NB * (NB + 1) / 2)       // 2080 upper-triangular tile pairs

// LDS B-tile: 128 rows x 64 halfs, padded to 72 halfs (144 B = 36 words) per row
// so 16 consecutive rows map to disjoint bank groups (36*l mod 64 covers all
// 16 multiples of 4 -> conflict-free 16B lane accesses).
#define BT_PITCH 72

// d_ws layout:
//   [0, XN_BYTES)        : x_n in f16, row-major [8192][64]
//   [XN_BYTES, +67*4)    : float accum: [0]=rep_sum [1]=sum_norm [2]=sum_norm2 [3..66]=colsum[64]
#define XN_BYTES (BATCH * DIM * 2)
#define N_ACCUM  67

// ---------------------------------------------------------------- init
__global__ void vmf_init_kernel(float* accum) {
    int t = threadIdx.x;
    if (t < N_ACCUM) accum[t] = 0.0f;
}

// ---------------------------------------------------------------- normalize
// 256 threads = 8 waves; one wave per row; lane handles 2 elements.
__global__ void __launch_bounds__(256) vmf_normalize_kernel(
    const float* __restrict__ x, _Float16* __restrict__ xn, float* __restrict__ accum) {
    __shared__ float lds_col[DIM];
    __shared__ float lds_s[2];

    int tid  = threadIdx.x;
    int w    = tid >> 5;
    int lane = tid & 31;
    if (tid < DIM) lds_col[tid] = 0.0f;
    if (tid < 2)   lds_s[tid]   = 0.0f;
    __syncthreads();

    int row = blockIdx.x * 8 + w;
    const float* xr = x + (size_t)row * DIM;
    float2 v = *(const float2*)(xr + lane * 2);
    float p = v.x * v.x + v.y * v.y;
    #pragma unroll
    for (int off = 16; off > 0; off >>= 1) p += __shfl_xor(p, off, 32);

    float norm = sqrtf(p);
    float inv  = 1.0f / fmaxf(norm, 1e-12f);
    float n0 = v.x * inv, n1 = v.y * inv;

    v2h hv; hv.x = (_Float16)n0; hv.y = (_Float16)n1;
    *(v2h*)(xn + (size_t)row * DIM + lane * 2) = hv;

    atomicAdd(&lds_col[lane * 2 + 0], n0);
    atomicAdd(&lds_col[lane * 2 + 1], n1);
    if (lane == 0) {
        atomicAdd(&lds_s[0], norm);
        atomicAdd(&lds_s[1], norm * norm);
    }
    __syncthreads();

    if (tid < DIM) atomicAdd(&accum[3 + tid], lds_col[tid]);
    if (tid == 64) atomicAdd(&accum[1], lds_s[0]);
    if (tid == 65) atomicAdd(&accum[2], lds_s[1]);
}

// ---------------------------------------------------------------- Gram + relu^2 reduction
// One 128x128 tile of G = Xn * Xn^T per workgroup; only bi<=bj pairs (symmetry).
// B tile staged to LDS with CDNA5 async global->LDS copies; A direct from global.
__global__ void __launch_bounds__(256) vmf_gram_kernel(
    const _Float16* __restrict__ xn, float* __restrict__ accum) {
    __shared__ _Float16 bt[TILE * BT_PITCH];

    // triangular mapping blockIdx -> (bi, bj), bi <= bj
    int rem = blockIdx.x;
    int bi = 0, left = NB;
    while (rem >= left) { rem -= left; ++bi; --left; }
    int bj = bi + rem;

    int tid     = threadIdx.x;
    int w       = tid >> 5;
    int lane    = tid & 31;
    int halfSel = lane >> 4;     // 0: lanes 0-15, 1: lanes 16-31
    int lidx    = lane & 15;

    int r0    = bi * TILE + w * 16;   // this wave's row strip
    int cBase = bj * TILE;

    // ---- async-stage B tile (128 rows x 128 bytes) into LDS, padded rows ----
    {
        unsigned bt_base = (unsigned)(uintptr_t)(void*)&bt[0]; // low 32 bits == LDS offset
        unsigned long long gbase =
            (unsigned long long)(uintptr_t)(xn + (size_t)cBase * DIM);
        #pragma unroll
        for (int i = 0; i < 4; ++i) {
            int chunk = tid + 256 * i;            // 0..1023, 16B chunks
            int row   = chunk >> 3;               // 8 chunks per 128B row
            int coff  = (chunk & 7) * 16;         // byte offset in row
            unsigned lds_off = bt_base + (unsigned)(row * (BT_PITCH * 2) + coff);
            unsigned long long ga = gbase + (unsigned long long)(row * 128 + coff);
            asm volatile("global_load_async_to_lds_b128 %0, %1, off"
                         :: "v"(lds_off), "v"(ga)
                         : "memory");
        }
        asm volatile("s_wait_asynccnt 0x0" ::: "memory");
    }
    __syncthreads();

    v8f acc[8] = {};

    #pragma unroll
    for (int k0 = 0; k0 < DIM; k0 += 32) {
        // ---- A fragment (16x32, f16) from global: lane holds row (r0+lidx),
        //      halfs 0-7 = K[kb..kb+7], halfs 8-15 = K[kb+16..kb+23], kb = k0 + 8*halfSel
        int arow = r0 + lidx;
        int kb   = k0 + halfSel * 8;
        const _Float16* ap = xn + (size_t)arow * DIM + kb;
        v8h alo = *(const v8h*)ap;
        v8h ahi = *(const v8h*)(ap + 16);
        v16h a;
        #pragma unroll
        for (int i = 0; i < 8; ++i) { a[i] = alo[i]; a[8 + i] = ahi[i]; }

        #pragma unroll
        for (int jt = 0; jt < 8; ++jt) {
            // ---- B fragment (32x16) from LDS: lane holds column (tile row jt*16+lidx),
            //      halfs 0-7 = K[kb2..kb2+7], halfs 8-15 = K[kb2+16..kb2+23], kb2 = k0 + 8*halfSel
            int brow = jt * 16 + lidx;
            int kb2  = k0 + halfSel * 8;
            const _Float16* bp = &bt[brow * BT_PITCH + kb2];
            v8h blo = *(const v8h*)bp;
            v8h bhi = *(const v8h*)(bp + 16);
            v16h b;
            #pragma unroll
            for (int i = 0; i < 8; ++i) { b[i] = blo[i]; b[8 + i] = bhi[i]; }
            acc[jt] = __builtin_amdgcn_wmma_f32_16x16x32_f16(
                false, a, false, b, (short)0, acc[jt], false, false);
        }
    }

    // C layout: VGPR v -> M = v + 8*halfSel, N = lidx.
    // Diagonal (gi==gj) only possible when bi==bj, jt==w, M==N.
    float lsum  = 0.0f;
    float scale = (bi < bj) ? 2.0f : 1.0f;
    bool  diagT = (bi == bj);
    #pragma unroll
    for (int jt = 0; jt < 8; ++jt) {
        #pragma unroll
        for (int v = 0; v < 8; ++v) {
            float g = acc[jt][v];
            float r = g > 0.0f ? g : 0.0f;
            float c = r * r;
            if (diagT && jt == w && lidx == v + halfSel * 8) c = 0.0f;
            lsum += c * scale;
        }
    }
    #pragma unroll
    for (int off = 16; off > 0; off >>= 1) lsum += __shfl_xor(lsum, off, 32);
    if (lane == 0) atomicAdd(&accum[0], lsum);
}

// ---------------------------------------------------------------- final scalar assembly
__global__ void vmf_final_kernel(const float* __restrict__ mu,
                                 const float* __restrict__ kappa_p,
                                 const float* __restrict__ accum,
                                 float* __restrict__ out) {
    if (threadIdx.x != 0) return;
    const float Bf = (float)BATCH;
    const float d  = (float)DIM;
    const float nu = d * 0.5f - 1.0f;   // 31

    float kappa_c = fmaxf(kappa_p[0], 0.001f);

    // mu_n and mean dot via column sums
    float mn = 0.0f;
    for (int c = 0; c < DIM; ++c) mn += mu[c] * mu[c];
    mn = fmaxf(sqrtf(mn), 1e-12f);
    float meandot = 0.0f;
    for (int c = 0; c < DIM; ++c) meandot += (accum[3 + c] / Bf) * (mu[c] / mn);

    // log I_nu(kappa) ascending series, log-space (64 terms)
    float lhx = logf(kappa_c * 0.5f);
    float tmax = -3.4e38f;
    for (int k = 0; k < 64; ++k) {
        float kf = (float)k;
        float t = (nu + 2.0f * kf) * lhx - lgammaf(kf + 1.0f) - lgammaf(nu + kf + 1.0f);
        tmax = fmaxf(tmax, t);
    }
    float s = 0.0f;
    for (int k = 0; k < 64; ++k) {
        float kf = (float)k;
        float t = (nu + 2.0f * kf) * lhx - lgammaf(kf + 1.0f) - lgammaf(nu + kf + 1.0f);
        s += expf(t - tmax);
    }
    float log_iv = tmax + logf(s);

    // logaddexp(log_iv, log(eps))
    float leps = logf(1e-10f);
    float m  = fmaxf(log_iv, leps);
    float lb = m + log1pf(expf(fminf(log_iv, leps) - m));

    float log_C = (d * 0.5f - 1.0f) * logf(kappa_c)
                - (d * 0.5f) * logf(2.0f * 3.14159265358979323846f)
                - lb;

    float vmf = -(log_C + kappa_c * meandot);

    float mean_norm = accum[1] / Bf;
    float radius    = accum[2] / Bf - mean_norm * mean_norm;  // mean((n - mean)^2)
    float rad_term  = 0.8f * radius;
    float rep_term  = 0.2f * (accum[0] / (Bf * (Bf - 1.0f)));

    out[0] = vmf + rad_term + rep_term;
    out[1] = vmf;
    out[2] = rad_term;
    out[3] = rep_term;
}

// ---------------------------------------------------------------- launcher
extern "C" void kernel_launch(void* const* d_in, const int* in_sizes, int n_in,
                              void* d_out, int out_size, void* d_ws, size_t ws_size,
                              hipStream_t stream) {
    const float* x     = (const float*)d_in[0];
    const float* mu    = (const float*)d_in[1];
    const float* kappa = (const float*)d_in[2];
    float* out = (float*)d_out;

    _Float16* xn = (_Float16*)d_ws;
    float* accum = (float*)((char*)d_ws + XN_BYTES);

    vmf_init_kernel<<<1, 128, 0, stream>>>(accum);
    vmf_normalize_kernel<<<BATCH / 8, 256, 0, stream>>>(x, xn, accum);
    vmf_gram_kernel<<<NPAIR, 256, 0, stream>>>(xn, accum);
    vmf_final_kernel<<<1, 32, 0, stream>>>(mu, kappa, accum, out);
}
